// PGCN_g_twin_26388279066913
// MI455X (gfx1250) — matrix operand order repaired and checked by reading, not compiled, not run on hardware
//
#include <hip/hip_runtime.h>

// ---------------------------------------------------------------------------
// Problem constants (fixed by the reference harness)
// ---------------------------------------------------------------------------
#define NNODES   64000      // B * NPG
#define NEDGES   512000     // B * EPG
#define FDIM     128
#define NGRAPH   64
#define NPG      1000
#define KTOP     500
#define GDIM3    384        // 3*F

typedef __attribute__((ext_vector_type(16))) __bf16 v16bf;
typedef __attribute__((ext_vector_type(8)))  __bf16 v8bf;
typedef __attribute__((ext_vector_type(8)))  float  v8f;

typedef __attribute__((ext_vector_type(4))) unsigned int u32x4;
typedef __attribute__((ext_vector_type(8))) int          i32x8;
typedef __attribute__((ext_vector_type(4))) int          i32x4;

#if defined(__has_builtin)
#if __has_builtin(__builtin_amdgcn_tensor_load_to_lds) && \
    __has_builtin(__builtin_amdgcn_s_wait_tensorcnt)
#define USE_TDM 1
#endif
#endif

static __device__ __forceinline__ __bf16 f2bf(float f) {
  union { float f; unsigned u; } in; in.f = f;
  unsigned u = in.u;
  unsigned r = (u + 0x7FFFu + ((u >> 16) & 1u)) >> 16;   // RNE
  union { unsigned short s; __bf16 b; } out; out.s = (unsigned short)r;
  return out.b;
}

// ---------------------------------------------------------------------------
// Utility kernels
// ---------------------------------------------------------------------------
__global__ void k_zero_f32(float* p, long n) {
  long i = (long)blockIdx.x * blockDim.x + threadIdx.x;
  if (i < n) p[i] = 0.0f;
}
__global__ void k_zero_i32(int* p, long n) {
  long i = (long)blockIdx.x * blockDim.x + threadIdx.x;
  if (i < n) p[i] = 0;
}

// fp32 -> bf16, 4 elements/thread
__global__ void k_cvt_bf16(const float* __restrict__ x, __bf16* __restrict__ y, long n) {
  long i = ((long)blockIdx.x * blockDim.x + threadIdx.x) * 4;
  if (i >= n) return;
  float4 v = *(const float4*)(x + i);
  union { __bf16 b[4]; int2 u; } o;
  o.b[0] = f2bf(v.x); o.b[1] = f2bf(v.y); o.b[2] = f2bf(v.z); o.b[3] = f2bf(v.w);
  *(int2*)(y + i) = o.u;
}

// ---------------------------------------------------------------------------
// Weight packing: W (128x128 f32, row-major [k][n]) -> bf16 fragments laid out
// so each lane's 16-element B-fragment (contiguous k order, same as the A fragment) is one
// contiguous 32-byte load.  packed[((kt*8+nt)*32 + lane)*16 + j] =
//   W[kt*32 + (lane>>4)*16 + j][nt*16 + (lane&15)]
// ---------------------------------------------------------------------------
__global__ void k_pack_w(const float* __restrict__ W, __bf16* __restrict__ Wpk) {
  int t = blockIdx.x * 256 + threadIdx.x;          // 16384 elements
  if (t >= 16384) return;
  int j    = t & 15;
  int lane = (t >> 4) & 31;
  int nt   = (t >> 9) & 7;
  int kt   = t >> 12;
  int k = kt * 32 + (lane >> 4) * 16 + j;
  int n = nt * 16 + (lane & 15);
  Wpk[t] = f2bf(W[k * FDIM + n]);
}

// ---------------------------------------------------------------------------
// GEMM: C[N,128] (f32) = A[N,128] (bf16) @ Wpk (bf16, pre-swizzled)
// Block = 256 threads = 8 waves; each wave computes a 32x128 strip:
// two A fragments share each B fragment (2 wmma per ds-load pair), K = 4 x 32.
// Weight tile (32 KB) staged into LDS by the Tensor Data Mover when available.
// ---------------------------------------------------------------------------
__global__ __launch_bounds__(256) void k_gemm_bf16(
    const __bf16* __restrict__ A, const __bf16* __restrict__ Wpk,
    float* __restrict__ C, int nrows)
{
  __shared__ __bf16 lw[4 * 8 * 32 * 16];   // 32 KB packed weight tile (LDS offset 0)

#ifdef USE_TDM
  if (threadIdx.x == 0) {
    // Tensor DMA descriptor (ISA 08_async_tensor §8): 1-D copy of 4096 x 8B
    // from Wpk to LDS offset 0. TDM ignores EXEC; wave 0 issues one op.
    unsigned long long ga = (unsigned long long)Wpk;
    u32x4 g0;
    g0[0] = 1u;                                          // count=1, user mode
    g0[1] = 0u;                                          // lds_addr = 0
    g0[2] = (unsigned)(ga & 0xFFFFFFFFull);              // global_addr[31:0]
    g0[3] = (unsigned)((ga >> 32) & 0x01FFFFFFull) | (2u << 30);  // addr[56:32] | type=2
    i32x8 g1;
    g1[0] = (int)(3u << 16);        // workgroup_mask=0, data_size=3 (8B)
    g1[1] = (int)(0x1000u << 16);   // tensor_dim0 low16 = 4096
    g1[2] = (int)(1u << 16);        // tensor_dim0 hi=0, tensor_dim1 low16 = 1
    g1[3] = (int)(4096u << 16);     // tensor_dim1 hi=0, tile_dim0 = 4096
    g1[4] = 1;                      // tile_dim1 = 1, tile_dim2 = 0
    g1[5] = 4096;                   // tensor_dim0_stride low32
    g1[6] = 0;                      // stride hi / tensor_dim1_stride lo
    g1[7] = 0;
    i32x4 z4 = {0, 0, 0, 0};
#if __clang_major__ >= 23
    i32x8 z8 = {0, 0, 0, 0, 0, 0, 0, 0};
    __builtin_amdgcn_tensor_load_to_lds(g0, g1, z4, z4, z8, 0);
#else
    __builtin_amdgcn_tensor_load_to_lds(g0, g1, z4, z4, 0);
#endif
    __builtin_amdgcn_s_wait_tensorcnt(0);
  }
#else
  {
    const int4* s = (const int4*)Wpk;
    int4* d = (int4*)lw;
    for (int i = threadIdx.x; i < 2048; i += 256) d[i] = s[i];
  }
#endif
  __syncthreads();

  // The TDM writes LDS through a descriptor the compiler cannot see: escape
  // the LDS pointer through opaque asm with a memory clobber so the compiler
  // must materialize the B-fragment ds_loads (prevents undef folding).
  const __bf16* lwp = lw;
#ifdef USE_TDM
  asm volatile("" : "+v"(lwp) :: "memory");
#endif

  const int wave = threadIdx.x >> 5, lane = threadIdx.x & 31;
  const long row0 = (long)blockIdx.x * 256 + wave * 32;   // 32 rows per wave
  if (row0 >= nrows) return;

  v8f acc0[8] = {};
  v8f acc1[8] = {};
  const int m  = lane & 15;
  const int kg = lane >> 4;                 // 0 or 1
  const __bf16* arow0 = A + (row0 + m) * FDIM;
  const __bf16* arow1 = A + (row0 + 16 + m) * FDIM;

  for (int kt = 0; kt < 4; ++kt) {
    // A fragment: contiguous k = 16*kg + j (matches packed B) -> two 16B chunks/lane
    v8bf lo0 = *(const v8bf*)(arow0 + kt * 32 +     kg * 16);
    v8bf hi0 = *(const v8bf*)(arow0 + kt * 32 + 8 + kg * 16);
    v8bf lo1 = *(const v8bf*)(arow1 + kt * 32 +     kg * 16);
    v8bf hi1 = *(const v8bf*)(arow1 + kt * 32 + 8 + kg * 16);
    v16bf afrag0, afrag1;
#pragma unroll
    for (int j = 0; j < 8; ++j) {
      afrag0[j] = lo0[j]; afrag0[j + 8] = hi0[j];
      afrag1[j] = lo1[j]; afrag1[j + 8] = hi1[j];
    }

#pragma unroll
    for (int nt = 0; nt < 8; ++nt) {
      v16bf bfrag = *(const v16bf*)(&lwp[(((kt * 8) + nt) * 32 + lane) * 16]);
      acc0[nt] = __builtin_amdgcn_wmma_f32_16x16x32_bf16(
          false, afrag0, false, bfrag, (short)0, acc0[nt], false, false);
      acc1[nt] = __builtin_amdgcn_wmma_f32_16x16x32_bf16(
          false, afrag1, false, bfrag, (short)0, acc1[nt], false, false);
    }
  }

  // C/D layout: VGPR r, lanes 0-15 -> M=r, lanes 16-31 -> M=8+r; N = nt*16+(lane&15)
  float* crow0 = C + row0 * FDIM;
  float* crow1 = C + (row0 + 16) * FDIM;
  const int nlo = lane & 15, mhi = (lane >> 4) * 8;
#pragma unroll
  for (int nt = 0; nt < 8; ++nt)
#pragma unroll
    for (int r = 0; r < 8; ++r) {
      crow0[(long)(mhi + r) * FDIM + nt * 16 + nlo] = acc0[nt][r];
      crow1[(long)(mhi + r) * FDIM + nt * 16 + nlo] = acc1[nt][r];
    }
}

// ---------------------------------------------------------------------------
// Degree accumulation (5 weight channels: edge_attr[:,0..3] and ones)
// then dinv = rsqrt(deg + 1)  (self-loop weight 1)
// ---------------------------------------------------------------------------
__global__ void k_degrees(const int* __restrict__ ei, const float* __restrict__ ea,
                          float* __restrict__ deg) {
  int e = blockIdx.x * 256 + threadIdx.x;
  if (e >= NEDGES) return;
  int col = ei[NEDGES + e];
  float4 w = *(const float4*)(ea + (long)e * 4);
  atomicAdd(&deg[0 * NNODES + col], w.x);
  atomicAdd(&deg[1 * NNODES + col], w.y);
  atomicAdd(&deg[2 * NNODES + col], w.z);
  atomicAdd(&deg[3 * NNODES + col], w.w);
  atomicAdd(&deg[4 * NNODES + col], 1.0f);
}
__global__ void k_make_dinv(float* deg, long n) {
  long i = (long)blockIdx.x * 256 + threadIdx.x;
  if (i < n) deg[i] = rsqrtf(deg[i] + 1.0f);   // deg+selfloop >= 1 > 0
}

// ---------------------------------------------------------------------------
// CSR build (sorted by destination node), reused by all 6 convs + score agg
// ---------------------------------------------------------------------------
__global__ void k_count_indeg(const int* __restrict__ ei, int* __restrict__ indeg) {
  int e = blockIdx.x * 256 + threadIdx.x;
  if (e >= NEDGES) return;
  atomicAdd(&indeg[ei[NEDGES + e]], 1);
}

// single-block exclusive scan over NNODES, writes offs[0..N] and rewrites
// cursor[i] (which held indeg) to the exclusive offset for the fill pass.
__global__ void k_scan(int* __restrict__ cursor, int* __restrict__ offs, int n) {
  __shared__ int sh[256];
  __shared__ int carry_s;
  int tid = threadIdx.x;
  if (tid == 0) carry_s = 0;
  __syncthreads();
  for (int base = 0; base < n; base += 256) {
    int i = base + tid;
    int v = (i < n) ? cursor[i] : 0;
    sh[tid] = v;
    __syncthreads();
    for (int off = 1; off < 256; off <<= 1) {
      int t = (tid >= off) ? sh[tid - off] : 0;
      __syncthreads();
      sh[tid] += t;
      __syncthreads();
    }
    int incl  = sh[tid];
    int carry = carry_s;
    if (i < n) { int excl = carry + incl - v; offs[i] = excl; cursor[i] = excl; }
    __syncthreads();
    if (tid == 255) carry_s = carry + incl;
    __syncthreads();
  }
  if (tid == 0) offs[n] = carry_s;
}

__global__ void k_fill_csr(const int* __restrict__ ei, int* __restrict__ cursor,
                           int* __restrict__ csr_src, int* __restrict__ csr_eid) {
  int e = blockIdx.x * 256 + threadIdx.x;
  if (e >= NEDGES) return;
  int row = ei[e], col = ei[NEDGES + e];
  int pos = atomicAdd(&cursor[col], 1);
  csr_src[pos] = row;
  csr_eid[pos] = e;
}

// ---------------------------------------------------------------------------
// Normalized gather (GCN aggregation) + bias + optional relu/scale/accumulate.
// One wave per node; lane covers 4 features (float4).  H rows are L2-resident
// (32.8 MB << 192 MB L2), so gather reads mostly hit L2.
//   y_i = dinv_i * ( sum_e dinv_src*w_e*H_src  +  dinv_i*H_i )
//   out_i (+)= scale * [relu](y_i + bias)
// ---------------------------------------------------------------------------
__global__ __launch_bounds__(256) void k_gather_conv(
    const float* __restrict__ H, const float* __restrict__ dinv,
    const int* __restrict__ offs, const int* __restrict__ csr_src,
    const int* __restrict__ csr_eid, const float* __restrict__ ea, int wcol,
    const float* __restrict__ bias, float* __restrict__ out,
    float scale, int accumulate, int do_relu)
{
  const int wave = threadIdx.x >> 5, lane = threadIdx.x & 31;
  const int node = blockIdx.x * 8 + wave;
  if (node >= NNODES) return;

  const int beg = offs[node], end = offs[node + 1];
  const float di = dinv[node];
  float a0 = 0.f, a1 = 0.f, a2 = 0.f, a3 = 0.f;

  for (int e = beg; e < end; ++e) {
    int src = csr_src[e];
    if (e + 1 < end)  // hint the next gathered row toward the caches
      __builtin_prefetch(H + (long)csr_src[e + 1] * FDIM + lane * 4, 0, 0);
    float w = (wcol >= 0) ? ea[(long)csr_eid[e] * 4 + wcol] : 1.0f;
    float c = dinv[src] * w;
    float4 hv = *(const float4*)(H + (long)src * FDIM + lane * 4);
    a0 += c * hv.x; a1 += c * hv.y; a2 += c * hv.z; a3 += c * hv.w;
  }
  float4 hs = *(const float4*)(H + (long)node * FDIM + lane * 4);
  a0 = di * (a0 + di * hs.x); a1 = di * (a1 + di * hs.y);
  a2 = di * (a2 + di * hs.z); a3 = di * (a3 + di * hs.w);

  float4 bv = *(const float4*)(bias + lane * 4);
  a0 += bv.x; a1 += bv.y; a2 += bv.z; a3 += bv.w;
  if (do_relu) {
    a0 = fmaxf(a0, 0.f); a1 = fmaxf(a1, 0.f);
    a2 = fmaxf(a2, 0.f); a3 = fmaxf(a3, 0.f);
  }
  a0 *= scale; a1 *= scale; a2 *= scale; a3 *= scale;

  float* orow = out + (long)node * FDIM + lane * 4;
  if (accumulate) {
    float4 pv = *(const float4*)orow;
    a0 += pv.x; a1 += pv.y; a2 += pv.z; a3 += pv.w;
  }
  float4 res; res.x = a0; res.y = a1; res.z = a2; res.w = a3;
  *(float4*)orow = res;
}

// ---------------------------------------------------------------------------
// Score head:  p[i] = g_i . Ws_rel,  q[i] = g_i . Ws_root   (g_i = [G1,G2,G3])
// then score[i] = bs + q[i] + sum_{incoming e} p[src]
// ---------------------------------------------------------------------------
__global__ __launch_bounds__(256) void k_score_pq(
    const float* __restrict__ G1, const float* __restrict__ G2,
    const float* __restrict__ G3, const float* __restrict__ Wrel,
    const float* __restrict__ Wroot, float* __restrict__ p, float* __restrict__ q)
{
  const int wave = threadIdx.x >> 5, lane = threadIdx.x & 31;
  const int node = blockIdx.x * 8 + wave;
  if (node >= NNODES) return;
  float sp = 0.f, sq = 0.f;
  for (int f = lane; f < GDIM3; f += 32) {
    float g = (f < 128) ? G1[(long)node * FDIM + f]
            : (f < 256) ? G2[(long)node * FDIM + (f - 128)]
                        : G3[(long)node * FDIM + (f - 256)];
    sp += g * Wrel[f];
    sq += g * Wroot[f];
  }
#pragma unroll
  for (int off = 16; off > 0; off >>= 1) {
    sp += __shfl_xor(sp, off);
    sq += __shfl_xor(sq, off);
  }
  if (lane == 0) { p[node] = sp; q[node] = sq; }
}

__global__ void k_score_combine(const int* __restrict__ offs,
                                const int* __restrict__ csr_src,
                                const float* __restrict__ p,
                                const float* __restrict__ q,
                                const float* __restrict__ bs,
                                float* __restrict__ score)
{
  int i = blockIdx.x * 256 + threadIdx.x;
  if (i >= NNODES) return;
  float s = q[i] + bs[0];
  int end = offs[i + 1];
  for (int e = offs[i]; e < end; ++e) s += p[csr_src[e]];
  score[i] = s;
}

// ---------------------------------------------------------------------------
// Per-graph exact top-k (rank counting, top_k tie semantics: earlier index
// wins on equal value) + tanh-weighted mean pool into r[graph][rcol..rcol+383]
// ---------------------------------------------------------------------------
__global__ __launch_bounds__(256) void k_topk_pool(
    const float* __restrict__ score, const float* __restrict__ G1,
    const float* __restrict__ G2, const float* __restrict__ G3,
    float* __restrict__ r, int rcol)
{
  __shared__ float ssc[NPG];
  __shared__ float acc[GDIM3];
  const int g = blockIdx.x, tid = threadIdx.x;
  for (int i = tid; i < NPG; i += 256) ssc[i] = score[(long)g * NPG + i];
  for (int i = tid; i < GDIM3; i += 256) acc[i] = 0.f;
  __syncthreads();

  for (int i = tid; i < NPG; i += 256) {
    float s = ssc[i];
    int rank = 0;
    for (int j = 0; j < NPG; ++j) {
      float t = ssc[j];
      rank += (t > s) || (t == s && j < i);
    }
    if (rank < KTOP) {
      float w = tanhf(s) * (1.0f / (float)KTOP);
      long node = (long)g * NPG + i;
      for (int f = 0; f < FDIM; ++f) atomicAdd(&acc[f],       w * G1[node * FDIM + f]);
      for (int f = 0; f < FDIM; ++f) atomicAdd(&acc[128 + f], w * G2[node * FDIM + f]);
      for (int f = 0; f < FDIM; ++f) atomicAdd(&acc[256 + f], w * G3[node * FDIM + f]);
    }
  }
  __syncthreads();
  for (int f = tid; f < GDIM3; f += 256) r[(long)g * (2 * GDIM3) + rcol + f] = acc[f];
}

// ---------------------------------------------------------------------------
// Final MLP: [64,768] -> 32 -> 8 -> 2, single block
// ---------------------------------------------------------------------------
__global__ __launch_bounds__(256) void k_mlp(
    const float* __restrict__ r,
    const float* __restrict__ Wm1, const float* __restrict__ bm1,
    const float* __restrict__ Wm2, const float* __restrict__ bm2,
    const float* __restrict__ Wm3, const float* __restrict__ bm3,
    float* __restrict__ out)
{
  __shared__ float h1[NGRAPH * 32];
  __shared__ float h2[NGRAPH * 8];
  const int tid = threadIdx.x;
  for (int o = tid; o < NGRAPH * 32; o += 256) {
    int b = o >> 5, c = o & 31;
    float s = bm1[c];
    for (int f = 0; f < 2 * GDIM3; ++f) s += r[(long)b * (2 * GDIM3) + f] * Wm1[f * 32 + c];
    h1[o] = fmaxf(s, 0.f);
  }
  __syncthreads();
  for (int o = tid; o < NGRAPH * 8; o += 256) {
    int b = o >> 3, c = o & 7;
    float s = bm2[c];
    for (int f = 0; f < 32; ++f) s += h1[b * 32 + f] * Wm2[f * 8 + c];
    h2[o] = fmaxf(s, 0.f);
  }
  __syncthreads();
  for (int o = tid; o < NGRAPH * 2; o += 256) {
    int b = o >> 1, c = o & 1;
    float s = bm3[c];
    for (int f = 0; f < 8; ++f) s += h2[b * 8 + f] * Wm3[f * 2 + c];
    out[o] = s;
  }
}

// ---------------------------------------------------------------------------
// Host orchestration
// ---------------------------------------------------------------------------
extern "C" void kernel_launch(void* const* d_in, const int* in_sizes, int n_in,
                              void* d_out, int out_size, void* d_ws, size_t ws_size,
                              hipStream_t stream)
{
  (void)in_sizes; (void)n_in; (void)out_size; (void)ws_size;

  const float* x[2]  = { (const float*)d_in[0], (const float*)d_in[3] };
  const float* ea[2] = { (const float*)d_in[1], (const float*)d_in[4] };
  const int*   ei[2] = { (const int*)d_in[2],   (const int*)d_in[5]   };
  const float* W[6]  = { (const float*)d_in[9],  (const float*)d_in[11],
                         (const float*)d_in[13], (const float*)d_in[15],
                         (const float*)d_in[17], (const float*)d_in[19] };
  const float* bw[6] = { (const float*)d_in[10], (const float*)d_in[12],
                         (const float*)d_in[14], (const float*)d_in[16],
                         (const float*)d_in[18], (const float*)d_in[20] };
  const float* Ws_rel  = (const float*)d_in[21];
  const float* bs_rel  = (const float*)d_in[22];
  const float* Ws_root = (const float*)d_in[23];
  const float* Wm1 = (const float*)d_in[24]; const float* bm1 = (const float*)d_in[25];
  const float* Wm2 = (const float*)d_in[26]; const float* bm2 = (const float*)d_in[27];
  const float* Wm3 = (const float*)d_in[28]; const float* bm3 = (const float*)d_in[29];

  // workspace carve-up (256B aligned)
  char* base = (char*)d_ws;
  size_t off = 0;
  auto alloc = [&](size_t bytes) -> void* {
    void* p = base + off;
    off = (off + bytes + 255) & ~(size_t)255;
    return p;
  };
  const size_t NF = (size_t)NNODES * FDIM;
  float*  H     = (float*)alloc(NF * 4);
  float*  G1    = (float*)alloc(NF * 4);
  float*  G2    = (float*)alloc(NF * 4);
  float*  G3    = (float*)alloc(NF * 4);
  __bf16* Abf   = (__bf16*)alloc(NF * 2);
  __bf16* Wpk   = (__bf16*)alloc((size_t)6 * 16384 * 2);
  float*  dinv  = (float*)alloc((size_t)5 * NNODES * 4);
  float*  pbuf  = (float*)alloc((size_t)NNODES * 4);
  float*  qbuf  = (float*)alloc((size_t)NNODES * 4);
  float*  score = (float*)alloc((size_t)NNODES * 4);
  int*    offs  = (int*)alloc((size_t)(NNODES + 1) * 4);
  int*    cursor= (int*)alloc((size_t)NNODES * 4);
  int*    csrs  = (int*)alloc((size_t)NEDGES * 4);
  int*    csre  = (int*)alloc((size_t)NEDGES * 4);
  float*  rbuf  = (float*)alloc((size_t)NGRAPH * 2 * GDIM3 * 4);

  const int EB = (NEDGES + 255) / 256;          // 2000
  const int NB = (NNODES + 255) / 256;          // 250
  const int GEMMB = NNODES / 256;               // 250 (32 rows/wave, 8 waves)
  const int NODEB = NNODES / 8;                 // 8000 (wave-per-node kernels)

  // pack all conv weights once (bf16, WMMA-B-fragment swizzled)
  for (int c = 0; c < 6; ++c)
    hipLaunchKernelGGL(k_pack_w, dim3(64), dim3(256), 0, stream, W[c], Wpk + (size_t)c * 16384);

  for (int br = 0; br < 2; ++br) {
    // --- degrees -> dinv (5 channels) ---
    hipLaunchKernelGGL(k_zero_f32, dim3((5 * NNODES + 255) / 256), dim3(256), 0, stream,
                       dinv, (long)5 * NNODES);
    hipLaunchKernelGGL(k_degrees, dim3(EB), dim3(256), 0, stream, ei[br], ea[br], dinv);
    hipLaunchKernelGGL(k_make_dinv, dim3((5 * NNODES + 255) / 256), dim3(256), 0, stream,
                       dinv, (long)5 * NNODES);

    // --- CSR (dst-sorted) ---
    hipLaunchKernelGGL(k_zero_i32, dim3(NB), dim3(256), 0, stream, cursor, (long)NNODES);
    hipLaunchKernelGGL(k_count_indeg, dim3(EB), dim3(256), 0, stream, ei[br], cursor);
    hipLaunchKernelGGL(k_scan, dim3(1), dim3(256), 0, stream, cursor, offs, NNODES);
    hipLaunchKernelGGL(k_fill_csr, dim3(EB), dim3(256), 0, stream, ei[br], cursor, csrs, csre);

    // --- stage 1: four parallel GCNConvs averaged into G1 ---
    hipLaunchKernelGGL(k_cvt_bf16, dim3((int)(NF / 4 + 255) / 256), dim3(256), 0, stream,
                       x[br], Abf, (long)NF);
    for (int k = 0; k < 4; ++k) {
      hipLaunchKernelGGL(k_gemm_bf16, dim3(GEMMB), dim3(256), 0, stream,
                         Abf, Wpk + (size_t)k * 16384, H, NNODES);
      hipLaunchKernelGGL(k_gather_conv, dim3(NODEB), dim3(256), 0, stream,
                         H, dinv + (size_t)k * NNODES, offs, csrs, csre, ea[br], k,
                         bw[k], G1, 0.25f, (k > 0) ? 1 : 0, 1);
    }

    // --- stage 2: G2 = relu(conv(G1)) ---
    hipLaunchKernelGGL(k_cvt_bf16, dim3((int)(NF / 4 + 255) / 256), dim3(256), 0, stream,
                       G1, Abf, (long)NF);
    hipLaunchKernelGGL(k_gemm_bf16, dim3(GEMMB), dim3(256), 0, stream,
                       Abf, Wpk + (size_t)4 * 16384, H, NNODES);
    hipLaunchKernelGGL(k_gather_conv, dim3(NODEB), dim3(256), 0, stream,
                       H, dinv + (size_t)4 * NNODES, offs, csrs, csre, ea[br], -1,
                       bw[4], G2, 1.0f, 0, 1);

    // --- stage 3: G3 = conv(G2)  (no relu) ---
    hipLaunchKernelGGL(k_cvt_bf16, dim3((int)(NF / 4 + 255) / 256), dim3(256), 0, stream,
                       G2, Abf, (long)NF);
    hipLaunchKernelGGL(k_gemm_bf16, dim3(GEMMB), dim3(256), 0, stream,
                       Abf, Wpk + (size_t)5 * 16384, H, NNODES);
    hipLaunchKernelGGL(k_gather_conv, dim3(NODEB), dim3(256), 0, stream,
                       H, dinv + (size_t)4 * NNODES, offs, csrs, csre, ea[br], -1,
                       bw[5], G3, 1.0f, 0, 0);

    // --- SAGPool score + top-k weighted mean pool ---
    hipLaunchKernelGGL(k_score_pq, dim3(NODEB), dim3(256), 0, stream,
                       G1, G2, G3, Ws_rel, Ws_root, pbuf, qbuf);
    hipLaunchKernelGGL(k_score_combine, dim3(NB), dim3(256), 0, stream,
                       offs, csrs, pbuf, qbuf, bs_rel, score);
    hipLaunchKernelGGL(k_topk_pool, dim3(NGRAPH), dim3(256), 0, stream,
                       score, G1, G2, G3, rbuf, br * GDIM3);
  }

  hipLaunchKernelGGL(k_mlp, dim3(1), dim3(256), 0, stream,
                     rbuf, Wm1, bm1, Wm2, bm2, Wm3, bm3, (float*)d_out);
}
